// GCM_64209761075490
// MI455X (gfx1250) — compile-verified, matrix operand order checked by reading, not run on hardware
//
#include <hip/hip_runtime.h>
#include <hip/hip_bf16.h>
#include <math.h>

typedef __attribute__((ext_vector_type(16))) __bf16 v16bf;
typedef __attribute__((ext_vector_type(8)))  __bf16 v8bf;
typedef __attribute__((ext_vector_type(8)))  float  v8f;

#define B_    32
#define L1_   512
#define L2_   16
#define D_    300
#define C_    256
#define E_    556
#define NCLS_ 3
#define DPAD  320      // 300 -> pad to mult of 32 (K dim)
#define EPAD  576      // 556 -> pad to mult of 32 (conv2 K dim)
#define NEP   576      // 556 -> pad to mult of 64 (attn GEMM N dim)
#define MR    (B_*L1_) // 16384 context rows

// ---------------------------------------------------------------------------
// WMMA fragment loader: 16-bit A/B layout (ISA 7.12.2).
// lane&15 selects row (A) / col (B); (lane>>4)*8 selects the K sub-block.
// VGPR0..3 hold K = kb+0..7, VGPR4..7 hold K = 16+kb+0..7 (2 bf16/VGPR).
// Rows are 16B-aligned since K is a multiple of 32.
// ---------------------------------------------------------------------------
__device__ __forceinline__ v16bf frag_load(const __bf16* p, int lane) {
  const int kb = (lane >> 4) << 3;
  v8bf lo = *reinterpret_cast<const v8bf*>(p + kb);
  v8bf hi = *reinterpret_cast<const v8bf*>(p + 16 + kb);
  v16bf f;
#pragma unroll
  for (int i = 0; i < 8; ++i) { f[i] = lo[i]; f[i + 8] = hi[i]; }
  return f;
}

__device__ __forceinline__ float apply_act(float v, int act) {
  if (act == 1) return fmaxf(v, 0.f);
  if (act == 2) return tanhf(v);
  return v;
}

// ---------------------------------------------------------------------------
// bf16 WMMA GEMM, 32x64 register tile per wave (2 A frags x 4 B frags ->
// 8 WMMAs per 32-deep K step; every fragment reused 2-4x).
// Y[M][N](f32) = act(X[M][K] @ W[N][K]^T + bias). M%32==0, N%64==0, K%32==0.
// ---------------------------------------------------------------------------
__global__ __launch_bounds__(256)
void k_gemm_bf16(const __bf16* __restrict__ X, const __bf16* __restrict__ W,
                 const float* __restrict__ bias, float* __restrict__ Y,
                 int M, int N, int K, int act) {
  const int lane = threadIdx.x;
  const int wid  = blockIdx.x * blockDim.y + threadIdx.y;
  const int tilesN = N >> 6;
  if (wid >= (M >> 5) * tilesN) return;   // wave-uniform exit
  const int tm = wid / tilesN, tn = wid % tilesN;

  const __bf16* xr0 = X + (size_t)(tm * 32 +      (lane & 15)) * K;
  const __bf16* xr1 = X + (size_t)(tm * 32 + 16 + (lane & 15)) * K;
  const __bf16* wr  = W + (size_t)(tn * 64 +      (lane & 15)) * K;
  const size_t wstep = (size_t)16 * K;

  v8f acc[8];
#pragma unroll
  for (int i = 0; i < 8; ++i) { v8f z = {}; acc[i] = z; }

  for (int k0 = 0; k0 < K; k0 += 32) {
    v16bf a0 = frag_load(xr0 + k0, lane);
    v16bf a1 = frag_load(xr1 + k0, lane);
#pragma unroll
    for (int j = 0; j < 4; ++j) {
      v16bf b = frag_load(wr + j * wstep + k0, lane);
      acc[j]     = __builtin_amdgcn_wmma_f32_16x16x32_bf16(false, a0, false, b,
                                                           (short)0, acc[j],     false, false);
      acc[4 + j] = __builtin_amdgcn_wmma_f32_16x16x32_bf16(false, a1, false, b,
                                                           (short)0, acc[4 + j], false, false);
    }
  }

  const int colb  = tn * 64 + (lane & 15);
  const int rhalf = (lane >> 4) << 3;
#pragma unroll
  for (int cb = 0; cb < 4; ++cb) {
    const int col = colb + cb * 16;
    const float bv = bias ? bias[col] : 0.f;
#pragma unroll
    for (int i = 0; i < 2; ++i) {
      const int rbase = tm * 32 + i * 16 + rhalf;
      v8f a = acc[i * 4 + cb];
#pragma unroll
      for (int j = 0; j < 8; ++j)
        Y[(size_t)(rbase + j) * N + col] = apply_act(a[j] + bv, act);
    }
  }
}

// ---------------------------------------------------------------------------
// conv1d (width 3, pad 1) as 3 accumulated WMMA GEMM passes, 32x64 wave tile.
// X rows are (b*L + l); shift rows by delta=kk-1, zero A-frags at l boundaries
// (data-select only; EXEC stays all-1s at the WMMA). L power of two, 32|L.
// ---------------------------------------------------------------------------
__global__ __launch_bounds__(256)
void k_convgemm_bf16(const __bf16* __restrict__ X, const __bf16* __restrict__ W3,
                     const float* __restrict__ bias, float* __restrict__ Y,
                     int M, int N, int K, int L, int act) {
  const int lane = threadIdx.x;
  const int wid  = blockIdx.x * blockDim.y + threadIdx.y;
  const int tilesN = N >> 6;
  if (wid >= (M >> 5) * tilesN) return;
  const int tm = wid / tilesN, tn = wid % tilesN;

  const int r0 = tm * 32 +      (lane & 15);
  const int r1 = tm * 32 + 16 + (lane & 15);
  const int l0 = r0 & (L - 1);
  const int l1 = r1 & (L - 1);
  const __bf16* wr = W3 + (size_t)(tn * 64 + (lane & 15)) * K;
  const size_t wstep = (size_t)16 * K;

  v8f acc[8];
#pragma unroll
  for (int i = 0; i < 8; ++i) { v8f z = {}; acc[i] = z; }

  for (int kk = 0; kk < 3; ++kk) {
    const int delta = kk - 1;
    const bool v0 = (unsigned)(l0 + delta) < (unsigned)L;
    const bool v1 = (unsigned)(l1 + delta) < (unsigned)L;
    const __bf16* x0 = X + (size_t)(r0 + delta) * K;
    const __bf16* x1 = X + (size_t)(r1 + delta) * K;
    const __bf16* wk = wr + (size_t)kk * N * K;
    for (int k0 = 0; k0 < K; k0 += 32) {
      v16bf a0, a1;
      if (v0) a0 = frag_load(x0 + k0, lane); else { v16bf z = {}; a0 = z; }
      if (v1) a1 = frag_load(x1 + k0, lane); else { v16bf z = {}; a1 = z; }
#pragma unroll
      for (int j = 0; j < 4; ++j) {
        v16bf b = frag_load(wk + j * wstep + k0, lane);
        acc[j]     = __builtin_amdgcn_wmma_f32_16x16x32_bf16(false, a0, false, b,
                                                             (short)0, acc[j],     false, false);
        acc[4 + j] = __builtin_amdgcn_wmma_f32_16x16x32_bf16(false, a1, false, b,
                                                             (short)0, acc[4 + j], false, false);
      }
    }
  }

  const int colb  = tn * 64 + (lane & 15);
  const int rhalf = (lane >> 4) << 3;
#pragma unroll
  for (int cb = 0; cb < 4; ++cb) {
    const int col = colb + cb * 16;
    const float bv = bias ? bias[col] : 0.f;
#pragma unroll
    for (int i = 0; i < 2; ++i) {
      const int rbase = tm * 32 + i * 16 + rhalf;
      v8f a = acc[i * 4 + cb];
#pragma unroll
      for (int j = 0; j < 8; ++j)
        Y[(size_t)(rbase + j) * N + col] = apply_act(a[j] + bv, act);
    }
  }
}

// ---------------------------------------------------------------------------
// Embedding gather + bf16 convert, K-padded. grid=rows, block=DPAD.
// ---------------------------------------------------------------------------
__global__ void k_gather_bf16(const int* __restrict__ ids,
                              const float* __restrict__ emb,
                              __bf16* __restrict__ dst) {
  const int r = blockIdx.x, d = threadIdx.x;
  float v = 0.f;
  if (d < D_) v = emb[(size_t)ids[r] * D_ + d];
  dst[(size_t)r * DPAD + d] = (__bf16)v;
}

// Generic strided fp32 -> padded bf16 weight slab converter. grid=Ndst, block=Kdst.
__global__ void k_wconv(const float* __restrict__ src, __bf16* __restrict__ dst,
                        int Nsrc, int Ksrc, int Kdst, int rowStride, int colStride) {
  const int n = blockIdx.x, k = threadIdx.x;
  float v = 0.f;
  if (n < Nsrc && k < Ksrc) v = src[(size_t)n * rowStride + (size_t)k * colStride];
  dst[(size_t)n * Kdst + k] = (__bf16)v;
}

// conv3 over aspect (L2=16): a[b,m,o] = relu(sum_{i,kk} w[o,i,kk]*emb[ids[b,m+kk-1],i])
__global__ void k_conv3(const int* __restrict__ ids, const float* __restrict__ emb,
                        const float* __restrict__ w, const float* __restrict__ bias,
                        float* __restrict__ a_f32, __bf16* __restrict__ a_bf) {
  const int t = blockIdx.x * blockDim.x + threadIdx.x;   // B*L2*C threads
  const int o = t & (C_ - 1);
  const int bm = t >> 8;
  const int b = bm >> 4, m = bm & 15;
  float acc = bias[o];
  for (int kk = 0; kk < 3; ++kk) {
    const int mm = m + kk - 1;
    if ((unsigned)mm < (unsigned)L2_) {
      const float* e  = emb + (size_t)ids[b * L2_ + mm] * D_;
      const float* wp = w + (size_t)o * (D_ * 3) + kk;
      for (int i = 0; i < D_; ++i) acc += wp[i * 3] * e[i];
    }
  }
  acc = fmaxf(acc, 0.f);
  a_f32[t] = acc;
  a_bf[t]  = (__bf16)acc;
}

// attention: per context row, score over 16 aspect positions, softmax, weighted sum.
__global__ void k_attention(const float* __restrict__ preC, const float* __restrict__ preA,
                            const float* __restrict__ V, const float* __restrict__ a,
                            __bf16* __restrict__ att_bf) {
  __shared__ float lds[256];
  __shared__ float sc[16];
  __shared__ float alpha[16];
  const int row = blockIdx.x;          // 0..MR-1
  const int b   = row >> 9;            // /L1
  const int tid = threadIdx.x;
  const int mg = tid >> 4, j = tid & 15;

  const float* pc = preC + (size_t)row * NEP;
  const float* pa = preA + (size_t)(b * L2_ + mg) * NEP;
  float part = 0.f;
  for (int e = j; e < E_; e += 16) part += V[e] * tanhf(pc[e] + pa[e]);
  lds[tid] = part;
  __syncthreads();
  if (tid < 16) {
    float s = 0.f;
    for (int t = 0; t < 16; ++t) s += lds[tid * 16 + t];
    sc[tid] = s;
  }
  __syncthreads();
  if (tid == 0) {
    float mx = sc[0];
    for (int m = 1; m < 16; ++m) mx = fmaxf(mx, sc[m]);
    float sum = 0.f, ex[16];
    for (int m = 0; m < 16; ++m) { ex[m] = __expf(sc[m] - mx); sum += ex[m]; }
    for (int m = 0; m < 16; ++m) alpha[m] = ex[m] / sum;
  }
  __syncthreads();
  const float* ab = a + (size_t)b * L2_ * C_ + tid;
  float acc = 0.f;
  for (int m = 0; m < 16; ++m) acc += alpha[m] * ab[m * C_];
  att_bf[(size_t)row * C_ + tid] = (__bf16)acc;
}

// gin = concat(ctx[0:300], asp_lin[0:256], zeros pad) in bf16. grid=MR, block=EPAD.
__global__ void k_pack_gin(const __bf16* __restrict__ ctx_bf,
                           const float* __restrict__ asp_lin,
                           __bf16* __restrict__ gin) {
  const int r = blockIdx.x, c = threadIdx.x;
  __bf16 v;
  if (c < D_)        v = ctx_bf[(size_t)r * DPAD + c];
  else if (c < E_)   v = (__bf16)asp_lin[(size_t)r * C_ + (c - D_)];
  else               v = (__bf16)0.f;
  gin[(size_t)r * EPAD + c] = v;
}

// m = s1 * g (s1 already tanh'ed, g already relu'ed); emit f32 + bf16.
__global__ void k_mul(const float* __restrict__ s1, const float* __restrict__ g,
                      float* __restrict__ m, __bf16* __restrict__ m_bf) {
  const size_t i = (size_t)blockIdx.x * blockDim.x + threadIdx.x;
  const float v = s1[i] * g[i];
  m[i] = v;
  m_bf[i] = (__bf16)v;
}

// highway combine: mo = relu(hpre)*sig(gpre) + m*(1-sig(gpre)); write into hpre buf.
__global__ void k_combine(float* __restrict__ hpre, const float* __restrict__ gpre,
                          const float* __restrict__ m) {
  const size_t i = (size_t)blockIdx.x * blockDim.x + threadIdx.x;
  const float h  = fmaxf(hpre[i], 0.f);
  const float gt = 1.f / (1.f + __expf(-gpre[i]));
  hpre[i] = h * gt + m[i] * (1.f - gt);
}

// max-pool over L1 per (b,c). 8192 threads.
__global__ void k_pool(const float* __restrict__ mo, float* __restrict__ pooled) {
  const int t = blockIdx.x * blockDim.x + threadIdx.x;  // b*C + c
  const int b = t >> 8, c = t & (C_ - 1);
  const float* p = mo + (size_t)b * L1_ * C_ + c;
  float mx = p[0];
  for (int l = 1; l < L1_; ++l) mx = fmaxf(mx, p[(size_t)l * C_]);
  pooled[t] = mx;
}

// final classifier: out[b,k] = pooled[b] . out_w[k] + out_b[k]. 96 threads.
__global__ void k_out(const float* __restrict__ pooled, const float* __restrict__ ow,
                      const float* __restrict__ ob, float* __restrict__ out) {
  const int t = threadIdx.x;
  if (t >= B_ * NCLS_) return;
  const int b = t / NCLS_, k = t % NCLS_;
  float acc = ob[k];
  for (int c = 0; c < C_; ++c) acc += pooled[b * C_ + c] * ow[k * C_ + c];
  out[t] = acc;
}

// ---------------------------------------------------------------------------
extern "C" void kernel_launch(void* const* d_in, const int* in_sizes, int n_in,
                              void* d_out, int out_size, void* d_ws, size_t ws_size,
                              hipStream_t stream) {
  const int*   context_ids = (const int*)  d_in[0];
  const int*   aspect_ids  = (const int*)  d_in[1];
  const float* wordemb     = (const float*)d_in[2];
  const float* conv3_w     = (const float*)d_in[3];
  const float* conv3_b     = (const float*)d_in[4];
  const float* conv1_w     = (const float*)d_in[5];
  const float* conv1_b     = (const float*)d_in[6];
  const float* conv2_w     = (const float*)d_in[7];
  const float* conv2_b     = (const float*)d_in[8];
  const float* attn_W      = (const float*)d_in[9];
  const float* attn_V      = (const float*)d_in[10];
  const float* asp_w       = (const float*)d_in[11];
  const float* asp_b       = (const float*)d_in[12];
  const float* hw_w        = (const float*)d_in[13];
  const float* hw_b        = (const float*)d_in[14];
  const float* hwg_w       = (const float*)d_in[15];
  const float* hwg_b       = (const float*)d_in[16];
  const float* out_w       = (const float*)d_in[17];
  const float* out_b       = (const float*)d_in[18];
  float* out = (float*)d_out;

  char* ws = (char*)d_ws;
  size_t off = 0;
  auto alloc = [&](size_t bytes) -> void* {
    void* p = ws + off;
    off += (bytes + 255) & ~(size_t)255;
    return p;
  };

  __bf16* ctx_bf  = (__bf16*)alloc((size_t)MR * DPAD * 2);
  __bf16* W1b     = (__bf16*)alloc((size_t)NEP * DPAD * 2);
  __bf16* W2b     = (__bf16*)alloc((size_t)NEP * C_ * 2);
  __bf16* c1wb    = (__bf16*)alloc((size_t)3 * C_ * DPAD * 2);
  __bf16* c2wb    = (__bf16*)alloc((size_t)3 * C_ * EPAD * 2);
  __bf16* aspwb   = (__bf16*)alloc((size_t)C_ * C_ * 2);
  __bf16* hwwb    = (__bf16*)alloc((size_t)C_ * C_ * 2);
  __bf16* hwgwb   = (__bf16*)alloc((size_t)C_ * C_ * 2);
  float*  a_f32   = (float*) alloc((size_t)B_ * L2_ * C_ * 4);
  __bf16* a_bf    = (__bf16*)alloc((size_t)B_ * L2_ * C_ * 2);
  float*  preC    = (float*) alloc((size_t)MR * NEP * 4);
  float*  preA    = (float*) alloc((size_t)B_ * L2_ * NEP * 4);
  __bf16* att_bf  = (__bf16*)alloc((size_t)MR * C_ * 2);
  float*  asp_lin = (float*) alloc((size_t)MR * C_ * 4);
  __bf16* gin_bf  = (__bf16*)alloc((size_t)MR * EPAD * 2);
  float*  s1      = (float*) alloc((size_t)MR * C_ * 4);
  float*  g       = (float*) alloc((size_t)MR * C_ * 4);
  float*  m_f32   = (float*) alloc((size_t)MR * C_ * 4);
  __bf16* m_bf    = (__bf16*)alloc((size_t)MR * C_ * 2);
  float*  hpre    = (float*) alloc((size_t)MR * C_ * 4);
  float*  gpre    = (float*) alloc((size_t)MR * C_ * 4);
  float*  pooled  = (float*) alloc((size_t)B_ * C_ * 4);

  // 1) gather + convert context embeddings (K pad 300->320)
  k_gather_bf16<<<MR, DPAD, 0, stream>>>(context_ids, wordemb, ctx_bf);

  // 2) weight conversions to padded bf16 slabs
  k_wconv<<<NEP, DPAD, 0, stream>>>(attn_W,      W1b, E_, D_, DPAD, E_, 1);  // W1 = attn_W[:, :300]
  k_wconv<<<NEP, C_,   0, stream>>>(attn_W + D_, W2b, E_, C_, C_,   E_, 1);  // W2 = attn_W[:, 300:]
  for (int kk = 0; kk < 3; ++kk) {
    k_wconv<<<C_, DPAD, 0, stream>>>(conv1_w + kk, c1wb + (size_t)kk * C_ * DPAD,
                                     C_, D_, DPAD, D_ * 3, 3);
    k_wconv<<<C_, EPAD, 0, stream>>>(conv2_w + kk, c2wb + (size_t)kk * C_ * EPAD,
                                     C_, E_, EPAD, E_ * 3, 3);
  }
  k_wconv<<<C_, C_, 0, stream>>>(asp_w, aspwb, C_, C_, C_, C_, 1);
  k_wconv<<<C_, C_, 0, stream>>>(hw_w,  hwwb,  C_, C_, C_, C_, 1);
  k_wconv<<<C_, C_, 0, stream>>>(hwg_w, hwgwb, C_, C_, C_, C_, 1);

  // 3) aspect conv (small; scalar VALU)
  k_conv3<<<(B_ * L2_ * C_) / 256, 256, 0, stream>>>(aspect_ids, wordemb,
                                                     conv3_w, conv3_b, a_f32, a_bf);

  const dim3 gb(32, 8);
  auto blocks = [](int M, int N) { return ((M >> 5) * (N >> 6) + 7) / 8; };

  // 4) preC = ctx @ W1^T  (16384x576, K=320)
  k_gemm_bf16<<<blocks(MR, NEP), gb, 0, stream>>>(ctx_bf, W1b, nullptr, preC,
                                                  MR, NEP, DPAD, 0);
  // 5) preA = a @ W2^T    (512x576, K=256)
  k_gemm_bf16<<<blocks(B_ * L2_, NEP), gb, 0, stream>>>(a_bf, W2b, nullptr, preA,
                                                        B_ * L2_, NEP, C_, 0);
  // 6) attention: tanh-score, softmax over L2, weighted sum of a
  k_attention<<<MR, 256, 0, stream>>>(preC, preA, attn_V, a_f32, att_bf);

  // 7) asp_lin = att @ asp_w^T + asp_b
  k_gemm_bf16<<<blocks(MR, C_), gb, 0, stream>>>(att_bf, aspwb, asp_b, asp_lin,
                                                 MR, C_, C_, 0);
  // 8) gin = concat(ctx, asp_lin) bf16
  k_pack_gin<<<MR, EPAD, 0, stream>>>(ctx_bf, asp_lin, gin_bf);

  // 9) s1 = tanh(conv1(ctx)), 10) g = relu(conv2(gin))   — 3-tap WMMA convs
  k_convgemm_bf16<<<blocks(MR, C_), gb, 0, stream>>>(ctx_bf, c1wb, conv1_b, s1,
                                                     MR, C_, DPAD, L1_, 2);
  k_convgemm_bf16<<<blocks(MR, C_), gb, 0, stream>>>(gin_bf, c2wb, conv2_b, g,
                                                     MR, C_, EPAD, L1_, 1);
  // 11) m = s1 * g
  k_mul<<<MR, 256, 0, stream>>>(s1, g, m_f32, m_bf);

  // 12/13) highway pre-activations
  k_gemm_bf16<<<blocks(MR, C_), gb, 0, stream>>>(m_bf, hwwb,  hw_b,  hpre, MR, C_, C_, 0);
  k_gemm_bf16<<<blocks(MR, C_), gb, 0, stream>>>(m_bf, hwgwb, hwg_b, gpre, MR, C_, C_, 0);

  // 14) gate combine (result into hpre buffer)
  k_combine<<<MR, 256, 0, stream>>>(hpre, gpre, m_f32);

  // 15) max-pool over sequence, 16) classifier
  k_pool<<<(B_ * C_) / 256, 256, 0, stream>>>(hpre, pooled);
  k_out<<<1, 128, 0, stream>>>(pooled, out_w, out_b, out);

  (void)in_sizes; (void)n_in; (void)out_size; (void)ws_size;
}
